// curiosity_5892695130372
// MI455X (gfx1250) — matrix operand order repaired and checked by reading, not compiled, not run on hardware
//
#include <hip/hip_runtime.h>
#include <math.h>

typedef __bf16 bh;
typedef __attribute__((ext_vector_type(16))) __bf16 v16bf;
typedef __attribute__((ext_vector_type(8)))  __bf16 v8bf;
typedef __attribute__((ext_vector_type(8)))  float  v8f;

#define D_    1024
#define H_    16
#define DH_   64
#define B_    2
#define T_    2048
#define TA_   1500
#define TAP_  1504
#define D3_   3072
#define BT_   (B_ * T_)      // 4096
#define BTA_  (B_ * TA_)     // 3000
#define BTAP_ 3008           // padded rows for aux GEMM

// ---------------- WMMA helpers (CDNA5 wave32, 16x16x32 bf16) ----------------

__device__ __forceinline__ v8f wmma_bf16(v16bf a, v16bf b, v8f c) {
  return __builtin_amdgcn_wmma_f32_16x16x32_bf16(false, a, false, b, (short)0, c,
                                                 false, false);
}

// A fragment (16x32, row-major source): lane l holds row (l&15);
// elems 0..7 -> K = 8*half + i, elems 8..15 -> K = 16 + 8*half + i.
__device__ __forceinline__ v16bf load_a_frag(const bh* A, int lda, int row0,
                                             int k0, int lane) {
  int r = row0 + (lane & 15);
  int hf = (lane >> 4) & 1;
  const bh* p = A + (size_t)r * lda + k0 + 8 * hf;
  v8bf lo = *(const v8bf*)(p);
  v8bf hi = *(const v8bf*)(p + 16);
  return __builtin_shufflevector(lo, hi, 0, 1, 2, 3, 4, 5, 6, 7, 8, 9, 10, 11,
                                 12, 13, 14, 15);
}

// B fragment (32x16) from B^T stored row-major [N][K]: lane l holds col (l&15),
// elems 0..7 -> K = 8*half + i, elems 8..15 -> K = 16 + 8*half + i.
__device__ __forceinline__ v16bf load_bt_frag(const bh* BT, int ldb, int n0,
                                              int k0, int lane) {
  int n = n0 + (lane & 15);
  int hf = (lane >> 4) & 1;
  const bh* p = BT + (size_t)n * ldb + k0 + 8 * hf;
  v8bf lo = *(const v8bf*)(p);
  v8bf hi = *(const v8bf*)(p + 16);
  return __builtin_shufflevector(lo, hi, 0, 1, 2, 3, 4, 5, 6, 7, 8, 9, 10, 11,
                                 12, 13, 14, 15);
}

// 32x64 output tile per wave: 2 A-frags + 4 B-frags -> 8 WMMAs per k-step.
__device__ __forceinline__ void gemm_32x64(const bh* __restrict__ A,
                                           const bh* __restrict__ BT, int K,
                                           int lda, int ldb, int row0, int n0,
                                           int lane, v8f acc[2][4]) {
  v8f zero = {};
#pragma unroll
  for (int mt = 0; mt < 2; ++mt)
#pragma unroll
    for (int nt = 0; nt < 4; ++nt) acc[mt][nt] = zero;
  for (int k0 = 0; k0 < K; k0 += 32) {
    v16bf a0 = load_a_frag(A, lda, row0, k0, lane);
    v16bf a1 = load_a_frag(A, lda, row0 + 16, k0, lane);
#pragma unroll
    for (int nt = 0; nt < 4; ++nt) {
      v16bf b = load_bt_frag(BT, ldb, n0 + nt * 16, k0, lane);
      acc[0][nt] = wmma_bf16(a0, b, acc[0][nt]);
      acc[1][nt] = wmma_bf16(a1, b, acc[1][nt]);
    }
  }
}

// ---------------- async global -> LDS staging (gfx1250 ASYNCcnt path) -------

__device__ __forceinline__ void async_stage_b128(void* lds_dst,
                                                 const void* gsrc) {
  // Generic LDS pointer's low 32 bits are the wave-relative LDS address
  // (flat aperture drops the upper bits), which is exactly VDST for
  // GLOBAL_LOAD_ASYNC_TO_LDS.
  unsigned loff = (unsigned)(size_t)lds_dst;
  asm volatile("global_load_async_to_lds_b128 %0, %1, off"
               :
               : "v"(loff), "v"(gsrc)
               : "memory");
}

__device__ __forceinline__ void wait_async0() {
  asm volatile("s_wait_asynccnt 0x0" ::: "memory");
}

// ---------------- conversion / transpose pre-passes ----------------

__global__ void cvt_bf16_kernel(const float* __restrict__ in,
                                bh* __restrict__ out, int n_in, int n_out) {
  int i = blockIdx.x * blockDim.x + threadIdx.x;
  if (i < n_out) out[i] = (bh)(i < n_in ? in[i] : 0.f);
}

// in: [K][N] fp32 row-major  ->  out: [N][K] bf16 (i.e. W^T)
__global__ void cvt_transpose_kernel(const float* __restrict__ in,
                                     bh* __restrict__ out, int K, int N) {
  int i = blockIdx.x * blockDim.x + threadIdx.x;
  if (i < K * N) {
    int n = i / K;
    int kk = i - n * K;
    out[i] = (bh)in[(size_t)kk * N + n];
  }
}

// ---------------- QKV projection (main): C = x @ Wqkv + bqkv ----------------

__global__ __launch_bounds__(256) void qkv_main_kernel(
    const bh* __restrict__ Xb, const bh* __restrict__ WT,
    const float* __restrict__ bias, bh* __restrict__ q, bh* __restrict__ k,
    bh* __restrict__ vT) {
  int lane = threadIdx.x & 31, wave = threadIdx.x >> 5;
  int job = blockIdx.x * 8 + wave;
  const int colG = D3_ / 64;  // 48
  int tr = job / colG, tc = job - tr * colG;
  if (tr >= BT_ / 32) return;
  int row0 = tr * 32, n0 = tc * 64;
  v8f acc[2][4];
  gemm_32x64(Xb, WT, D_, D_, D_, row0, n0, lane, acc);
  int hf = lane >> 4, nn = lane & 15;
#pragma unroll
  for (int nt = 0; nt < 4; ++nt) {
    int c = n0 + nt * 16 + nn;
    float bv = bias[c];
#pragma unroll
    for (int mt = 0; mt < 2; ++mt)
#pragma unroll
      for (int r = 0; r < 8; ++r) {
        int row = row0 + mt * 16 + r + 8 * hf;
        int bb = row >> 11;
        int t = row & (T_ - 1);
        float val = acc[mt][nt][r] + bv;
        if (c < 1024) {
          int h = c >> 6, dh = c & 63;
          q[(((size_t)bb * H_ + h) * T_ + t) * DH_ + dh] = (bh)(val * 0.125f);
        } else if (c < 2048) {
          int cc = c - 1024;
          int h = cc >> 6, dh = cc & 63;
          k[(((size_t)bb * H_ + h) * T_ + t) * DH_ + dh] = (bh)val;
        } else {
          int cc = c - 2048;
          int h = cc >> 6, dh = cc & 63;
          vT[(((size_t)bb * H_ + h) * DH_ + dh) * T_ + t] = (bh)val;
        }
      }
  }
}

// ---------------- QKV projection (aux): only K and V columns ----------------

__global__ __launch_bounds__(256) void qkv_aux_kernel(
    const bh* __restrict__ XAb, const bh* __restrict__ WT,
    const float* __restrict__ bias, bh* __restrict__ ka,
    bh* __restrict__ vaT) {
  int lane = threadIdx.x & 31, wave = threadIdx.x >> 5;
  int job = blockIdx.x * 8 + wave;
  const int colG = 2048 / 64;  // 32 (cols 1024..3071)
  int tr = job / colG, tc = job - tr * colG;
  if (tr >= BTAP_ / 32) return;  // 94
  int row0 = tr * 32, n0 = 1024 + tc * 64;
  v8f acc[2][4];
  gemm_32x64(XAb, WT, D_, D_, D_, row0, n0, lane, acc);
  int hf = lane >> 4, nn = lane & 15;
#pragma unroll
  for (int nt = 0; nt < 4; ++nt) {
    int c = n0 + nt * 16 + nn;
    float bv = bias[c];
#pragma unroll
    for (int mt = 0; mt < 2; ++mt)
#pragma unroll
      for (int r = 0; r < 8; ++r) {
        int row = row0 + mt * 16 + r + 8 * hf;
        if (row >= BTA_) continue;
        int bb = (row >= TA_) ? 1 : 0;
        int t = row - bb * TA_;
        float val = acc[mt][nt][r] + bv;
        if (c < 2048) {
          int cc = c - 1024;
          int h = cc >> 6, dh = cc & 63;
          ka[(((size_t)bb * H_ + h) * TAP_ + t) * DH_ + dh] = (bh)val;
        } else {
          int cc = c - 2048;
          int h = cc >> 6, dh = cc & 63;
          vaT[(((size_t)bb * H_ + h) * DH_ + dh) * TAP_ + t] = (bh)val;
        }
      }
  }
}

// ---------------- block-cooperative flash attention ----------------
// 8 waves per block share one (b,h); K/V tiles (32 keys) staged into LDS via
// async global->LDS copies, double-buffered, one barrier per step.

__device__ __forceinline__ void stage_kv(bh* kdst, bh* vdst,
                                         const bh* __restrict__ Kbh,
                                         const bh* __restrict__ VTbh, int ldv,
                                         int kb, int tid) {
  {  // K tile: 32 keys x 64 dh, row-major (256 lanes x 16B)
    int key = tid >> 3, off = (tid & 7) * 8;
    async_stage_b128(kdst + key * DH_ + off,
                     Kbh + (size_t)(kb + key) * DH_ + off);
  }
  {  // V tile (transposed): 64 dh x 32 keys (256 lanes x 16B)
    int dh = tid >> 2, kk = (tid & 3) * 8;
    async_stage_b128(vdst + dh * 32 + kk,
                     VTbh + (size_t)dh * ldv + kb + kk);
  }
}

__device__ __forceinline__ void flash_pass(
    v16bf aq0, v16bf aq1, const bh* __restrict__ Kbh,
    const bh* __restrict__ VTbh, int ldv, int nkeys_pad, int nvalid,
    bh* kbuf0, bh* kbuf1, bh* vbuf0, bh* vbuf1, bh* myp, int lane, int tid,
    float m[8], float ssum[8], v8f o[4]) {
  int hf = lane >> 4, nn = lane & 15;
  bh* kB[2] = {kbuf0, kbuf1};
  bh* vB[2] = {vbuf0, vbuf1};
  v8f zero = {};
#pragma unroll
  for (int r = 0; r < 8; ++r) {
    m[r] = -__builtin_inff();
    ssum[r] = 0.f;
  }
#pragma unroll
  for (int t = 0; t < 4; ++t) o[t] = zero;

  stage_kv(kB[0], vB[0], Kbh, VTbh, ldv, 0, tid);
  int ibuf = 0;
  for (int kb = 0; kb < nkeys_pad; kb += 32, ibuf ^= 1) {
    wait_async0();    // this thread's async LDS writes landed
    __syncthreads();  // whole block's tile published; prev compute done
    if (kb + 32 < nkeys_pad)
      stage_kv(kB[ibuf ^ 1], vB[ibuf ^ 1], Kbh, VTbh, ldv, kb + 32, tid);
    if (kb + 64 < nkeys_pad) {
      __builtin_prefetch(Kbh + (size_t)(kb + 64) * DH_, 0, 1);
      __builtin_prefetch(VTbh + (size_t)(tid & 63) * ldv + kb + 64, 0, 1);
    }
    const bh* kt = kB[ibuf];
    const bh* vt = vB[ibuf];

    // S = (q*scale) @ K^T : 2 key-tiles x 2 dh-steps
    v8f s0 = zero, s1 = zero;
    s0 = wmma_bf16(aq0, load_bt_frag(kt, DH_, 0, 0, lane), s0);
    s0 = wmma_bf16(aq1, load_bt_frag(kt, DH_, 0, 32, lane), s0);
    s1 = wmma_bf16(aq0, load_bt_frag(kt, DH_, 16, 0, lane), s1);
    s1 = wmma_bf16(aq1, load_bt_frag(kt, DH_, 16, 32, lane), s1);

    bool v0 = (kb + nn) < nvalid;
    bool v1 = (kb + 16 + nn) < nvalid;

    float alpha[8];
#pragma unroll
    for (int r = 0; r < 8; ++r) {
      float a0 = v0 ? s0[r] : -__builtin_inff();
      float a1 = v1 ? s1[r] : -__builtin_inff();
      float mx = fmaxf(a0, a1);
      mx = fmaxf(mx, __shfl_xor(mx, 1));
      mx = fmaxf(mx, __shfl_xor(mx, 2));
      mx = fmaxf(mx, __shfl_xor(mx, 4));
      mx = fmaxf(mx, __shfl_xor(mx, 8));
      float mnew = fmaxf(m[r], mx);
      float al = __expf(m[r] - mnew);
      m[r] = mnew;
      float p0 = v0 ? __expf(a0 - mnew) : 0.f;
      float p1 = v1 ? __expf(a1 - mnew) : 0.f;
      float rs = p0 + p1;
      rs += __shfl_xor(rs, 1);
      rs += __shfl_xor(rs, 2);
      rs += __shfl_xor(rs, 4);
      rs += __shfl_xor(rs, 8);
      ssum[r] = ssum[r] * al + rs;
      alpha[r] = al;
      myp[(r + 8 * hf) * 32 + nn] = (bh)p0;
      myp[(r + 8 * hf) * 32 + 16 + nn] = (bh)p1;
    }
#pragma unroll
    for (int t = 0; t < 4; ++t)
#pragma unroll
      for (int r = 0; r < 8; ++r) o[t][r] *= alpha[r];

    // O += P @ V  (P bounced through per-wave LDS, V from staged tile)
    v16bf pf = load_a_frag(myp, 32, 0, 0, lane);
#pragma unroll
    for (int t = 0; t < 4; ++t) {
      v16bf bv = load_bt_frag(vt, 32, t * 16, 0, lane);
      o[t] = wmma_bf16(pf, bv, o[t]);
    }
  }
  __syncthreads();  // protect buffers before caller re-stages
}

__global__ __launch_bounds__(256) void attn_kernel(
    const bh* __restrict__ q, const bh* __restrict__ k,
    const bh* __restrict__ vT, const bh* __restrict__ ka,
    const bh* __restrict__ vaT, const float* __restrict__ g,
    bh* __restrict__ hc) {
  __shared__ bh kbuf[2][32 * DH_];
  __shared__ bh vbuf[2][DH_ * 32];
  __shared__ bh pbuf[8][16 * 32];
  int tid = threadIdx.x;
  int lane = tid & 31, wave = tid >> 5;
  int blk = blockIdx.x;        // 0..511
  int qblk = blk & 15;         // 16 q-blocks of 128 rows per (b,h)
  int h = (blk >> 4) & 15;
  int bb = blk >> 8;
  int q0 = qblk * 128 + wave * 16;
  const bh* Qbh = q + (((size_t)bb * H_ + h) * T_) * DH_;
  const bh* Kbh = k + (((size_t)bb * H_ + h) * T_) * DH_;
  const bh* VTbh = vT + (((size_t)bb * H_ + h) * DH_) * T_;
  const bh* KAbh = ka + (((size_t)bb * H_ + h) * TAP_) * DH_;
  const bh* VATbh = vaT + (((size_t)bb * H_ + h) * DH_) * TAP_;
  int hf = lane >> 4, nn = lane & 15;
  bh* myp = pbuf[wave];

  v16bf aq0 = load_a_frag(Qbh, DH_, q0, 0, lane);
  v16bf aq1 = load_a_frag(Qbh, DH_, q0, 32, lane);

  float m[8], ssum[8];
  v8f o[4];

  // self attention
  flash_pass(aq0, aq1, Kbh, VTbh, T_, T_, T_, kbuf[0], kbuf[1], vbuf[0],
             vbuf[1], myp, lane, tid, m, ssum, o);
  float hm[4][8];
#pragma unroll
  for (int t = 0; t < 4; ++t)
#pragma unroll
    for (int r = 0; r < 8; ++r) hm[t][r] = o[t][r] / ssum[r];

  // cross attention (keys padded 1500 -> 1504, masked)
  flash_pass(aq0, aq1, KAbh, VATbh, TAP_, TAP_, TA_, kbuf[0], kbuf[1], vbuf[0],
             vbuf[1], myp, lane, tid, m, ssum, o);

  float gate = 1.f / (1.f + __expf(-g[h]));
#pragma unroll
  for (int t = 0; t < 4; ++t)
#pragma unroll
    for (int r = 0; r < 8; ++r) {
      float ha = o[t][r] / ssum[r];
      float val = hm[t][r] * (1.f - gate) + ha * gate;
      int row = bb * T_ + q0 + r + 8 * hf;
      int col = h * 64 + t * 16 + nn;
      hc[(size_t)row * D_ + col] = (bh)val;
    }
}

// ---------------- output projection: out = hc @ Wo + bo (fp32 out) ----------

__global__ __launch_bounds__(256) void out_proj_kernel(
    const bh* __restrict__ A, const bh* __restrict__ WoT,
    const float* __restrict__ bo, float* __restrict__ out) {
  int lane = threadIdx.x & 31, wave = threadIdx.x >> 5;
  int job = blockIdx.x * 8 + wave;
  const int colG = D_ / 64;  // 16
  int tr = job / colG, tc = job - tr * colG;
  if (tr >= BT_ / 32) return;
  int row0 = tr * 32, n0 = tc * 64;
  v8f acc[2][4];
  gemm_32x64(A, WoT, D_, D_, D_, row0, n0, lane, acc);
  int hf = lane >> 4, nn = lane & 15;
#pragma unroll
  for (int nt = 0; nt < 4; ++nt) {
    int c = n0 + nt * 16 + nn;
    float bv = bo[c];
#pragma unroll
    for (int mt = 0; mt < 2; ++mt)
#pragma unroll
      for (int r = 0; r < 8; ++r) {
        int row = row0 + mt * 16 + r + 8 * hf;
        out[(size_t)row * D_ + c] = acc[mt][nt][r] + bv;
      }
  }
}

// ---------------- host orchestration ----------------

extern "C" void kernel_launch(void* const* d_in, const int* in_sizes, int n_in,
                              void* d_out, int out_size, void* d_ws,
                              size_t ws_size, hipStream_t stream) {
  const float* x = (const float*)d_in[0];
  const float* xa = (const float*)d_in[1];
  const float* Wqkv = (const float*)d_in[2];
  const float* bqkv = (const float*)d_in[3];
  const float* Wqkv_aux = (const float*)d_in[4];
  const float* bqkv_aux = (const float*)d_in[5];
  const float* Wo = (const float*)d_in[6];
  const float* bo = (const float*)d_in[7];
  const float* g = (const float*)d_in[8];
  float* out = (float*)d_out;
  (void)in_sizes; (void)n_in; (void)out_size; (void)ws_size;

  char* ws = (char*)d_ws;
  size_t off = 0;
  auto alloc_bh = [&](size_t elems) -> bh* {
    bh* p = (bh*)(ws + off);
    off += (elems * sizeof(bh) + 255) & ~(size_t)255;
    return p;
  };
  bh* xb = alloc_bh((size_t)BT_ * D_);
  bh* xab = alloc_bh((size_t)BTAP_ * D_);
  bh* WqkvT = alloc_bh((size_t)D3_ * D_);
  bh* WqkvAT = alloc_bh((size_t)D3_ * D_);
  bh* WoT = alloc_bh((size_t)D_ * D_);
  bh* qb = alloc_bh((size_t)B_ * H_ * T_ * DH_);
  bh* kb = alloc_bh((size_t)B_ * H_ * T_ * DH_);
  bh* vTb = alloc_bh((size_t)B_ * H_ * DH_ * T_);
  bh* kab = alloc_bh((size_t)B_ * H_ * TAP_ * DH_);
  bh* vaTb = alloc_bh((size_t)B_ * H_ * DH_ * TAP_);
  bh* hcb = alloc_bh((size_t)BT_ * D_);

  const int TB = 256;
  {
    int n = BT_ * D_;
    cvt_bf16_kernel<<<(n + TB - 1) / TB, TB, 0, stream>>>(x, xb, n, n);
  }
  {
    int nin = BTA_ * D_, nout = BTAP_ * D_;
    cvt_bf16_kernel<<<(nout + TB - 1) / TB, TB, 0, stream>>>(xa, xab, nin, nout);
  }
  {
    int n = D_ * D3_;
    cvt_transpose_kernel<<<(n + TB - 1) / TB, TB, 0, stream>>>(Wqkv, WqkvT, D_, D3_);
    cvt_transpose_kernel<<<(n + TB - 1) / TB, TB, 0, stream>>>(Wqkv_aux, WqkvAT, D_, D3_);
  }
  {
    int n = D_ * D_;
    cvt_transpose_kernel<<<(n + TB - 1) / TB, TB, 0, stream>>>(Wo, WoT, D_, D_);
  }
  {
    int jobs = (BT_ / 32) * (D3_ / 64);  // 6144
    qkv_main_kernel<<<jobs / 8, TB, 0, stream>>>(xb, WqkvT, bqkv, qb, kb, vTb);
  }
  {
    int jobs = (BTAP_ / 32) * (2048 / 64);  // 3008
    qkv_aux_kernel<<<jobs / 8, TB, 0, stream>>>(xab, WqkvAT, bqkv_aux, kab, vaTb);
  }
  {
    attn_kernel<<<B_ * H_ * (T_ / 128), TB, 0, stream>>>(qb, kb, vTb, kab,
                                                         vaTb, g, hcb);
  }
  {
    int jobs = (BT_ / 32) * (D_ / 64);  // 2048
    out_proj_kernel<<<jobs / 8, TB, 0, stream>>>(hcb, WoT, bo, out);
  }
}
